// GCN_15195594293582
// MI455X (gfx1250) — matrix operand order repaired and checked by reading, not compiled, not run on hardware
//
#include <hip/hip_runtime.h>
#include <hip/hip_bf16.h>
#include <math.h>

// ---------------- problem constants ----------------
#define N_NODE 2000
#define SPAD   2048            // K and N padded to 2048 (64 k-steps, 128 n-tiles)
#define DM     2048
#define INC    7
#define NDIM   40
#define CCH    64
#define KER    49
#define BATCHN 64
#define MROWS  (BATCHN * CCH)  // 4096 rows in the hop GEMMs

typedef __attribute__((ext_vector_type(16))) __bf16 v16bf;
typedef __attribute__((ext_vector_type(8)))  float  v8f;

union FragBF { v16bf v; unsigned int u[8]; };

static __device__ __forceinline__ unsigned short f2b(float f) {
  unsigned int u = __float_as_uint(f);
  u += 0x7fffu + ((u >> 16) & 1u);          // round-to-nearest-even
  return (unsigned short)(u >> 16);
}
static __device__ __forceinline__ float b2f(unsigned short h) {
  return __uint_as_float(((unsigned int)h) << 16);
}

// ---------------------------------------------------------------------------
// 1) adjacency: a = 0.5*softmax_row(relu(E1 @ E2)) + 0.5*I  (bf16, padded)
//    softmax rows sum to 1 => (adp+I)/rowsum == (adp+I)/2 exactly.
//    Rows [2000,2048) are zero so padded N-tiles produce exact zeros.
// ---------------------------------------------------------------------------
__global__ void k_adj(const float* __restrict__ nv1, const float* __restrict__ nv2,
                      unsigned short* __restrict__ a_bf) {
  const int v = blockIdx.x;
  const int t = threadIdx.x;
  if (v >= N_NODE) {                        // zero pad rows
    #pragma unroll
    for (int i = 0; i < 8; ++i) a_bf[(unsigned)v * SPAD + t + i * 256] = 0;
    return;
  }
  __shared__ float e1[NDIM];
  __shared__ float red[256];
  if (t < NDIM) e1[t] = nv1[v * NDIM + t];
  __syncthreads();

  float val[8];
  float mx = -1e30f;
  #pragma unroll
  for (int i = 0; i < 8; ++i) {
    const int w = t + i * 256;
    float s = 0.f;
    if (w < N_NODE) {
      #pragma unroll
      for (int d = 0; d < NDIM; ++d) s += e1[d] * nv2[d * N_NODE + w];
      s = fmaxf(s, 0.f);
      mx = fmaxf(mx, s);
    }
    val[i] = s;
  }
  red[t] = mx; __syncthreads();
  for (int st = 128; st > 0; st >>= 1) {
    if (t < st) red[t] = fmaxf(red[t], red[t + st]);
    __syncthreads();
  }
  mx = red[0]; __syncthreads();

  float sum = 0.f;
  #pragma unroll
  for (int i = 0; i < 8; ++i) {
    const int w = t + i * 256;
    if (w < N_NODE) { val[i] = expf(val[i] - mx); sum += val[i]; }
  }
  red[t] = sum; __syncthreads();
  for (int st = 128; st > 0; st >>= 1) {
    if (t < st) red[t] += red[t + st];
    __syncthreads();
  }
  const float inv = 1.f / red[0];

  #pragma unroll
  for (int i = 0; i < 8; ++i) {
    const int w = t + i * 256;
    float o = 0.f;
    if (w < N_NODE) o = 0.5f * val[i] * inv + (w == v ? 0.5f : 0.f);
    a_bf[(unsigned)v * SPAD + w] = f2b(o);   // pad columns -> 0
  }
}

// ---------------------------------------------------------------------------
// 2) start conv1d: (B,7,2048) * (64,7,49) -> (B,64,2000); f32 + bf16 copies
// ---------------------------------------------------------------------------
__global__ void k_conv(const float* __restrict__ x, const float* __restrict__ w_start,
                       const float* __restrict__ b_start,
                       float* __restrict__ outf, unsigned short* __restrict__ h0) {
  __shared__ float wc[INC * KER];
  const int c = blockIdx.y, b = blockIdx.z;
  const int t = threadIdx.x;
  for (int i = t; i < INC * KER; i += 256) wc[i] = w_start[c * INC * KER + i];
  __syncthreads();

  const int n = blockIdx.x * 256 + t;       // 0..2047
  const unsigned row = (unsigned)(b * CCH + c);
  float s = 0.f;
  if (n < N_NODE) {
    s = b_start[c];
    const float* xb = x + (size_t)b * (INC * DM);
    #pragma unroll
    for (int i = 0; i < INC; ++i) {
      const float* xi = xb + i * DM + n;
      const float* wi = wc + i * KER;
      for (int k = 0; k < KER; ++k) s += xi[k] * wi[k];
    }
  }
  outf[row * SPAD + n] = s;        // pad columns -> 0
  h0[row * SPAD + n]   = f2b(s);
}

// ---------------------------------------------------------------------------
// 3) mixprop hop (WMMA bf16): Hout = bf16( 0.05*out + 0.95*(Hin @ a^T) )
//    Register-blocked: one wave32 computes a 32x64 output block
//    (2 m-tiles x 4 n-tiles = 8 accumulators), 64 K-steps of 8 WMMAs.
//    A fragments reused across 4 n-tiles, B fragments across 2 m-tiles:
//    ~21 lane-FLOP per loaded byte from L2.
// ---------------------------------------------------------------------------
#define MT 2
#define NT 4
__global__ void __launch_bounds__(32)
k_hop(const unsigned short* __restrict__ Hin, const unsigned short* __restrict__ Abf,
      const float* __restrict__ outf, unsigned short* __restrict__ Hout) {
  const int ng = blockIdx.x;          // 0..31  (n-groups of NT tiles)
  const int mg = blockIdx.y;          // 0..127 (m-groups of MT tiles)
  const int lane = threadIdx.x;
  const int half = lane >> 4;
  const int lm   = lane & 15;
  const int m0 = mg * (MT * 16), n0 = ng * (NT * 16);

  const unsigned short* hrow[MT];
  const unsigned short* arow[NT];
  #pragma unroll
  for (int mi = 0; mi < MT; ++mi)
    hrow[mi] = Hin + (unsigned)(m0 + mi * 16 + lm) * SPAD;            // A: H[m,k]
  #pragma unroll
  for (int ni = 0; ni < NT; ++ni)
    arow[ni] = Abf + (unsigned)(n0 + ni * 16 + lm) * SPAD + half * 16; // B: a[n,k]

  v8f acc[MT][NT];
  #pragma unroll
  for (int mi = 0; mi < MT; ++mi)
    #pragma unroll
    for (int ni = 0; ni < NT; ++ni) acc[mi][ni] = (v8f){};

  for (int k0 = 0; k0 < SPAD; k0 += 32) {
    if (k0 + 64 < SPAD) {             // hint next K-slab toward L2
      __builtin_prefetch(hrow[0] + k0 + 64, 0, 0);
      __builtin_prefetch(hrow[1] + k0 + 64, 0, 0);
      __builtin_prefetch(arow[0] + k0 + 64, 0, 0);
      __builtin_prefetch(arow[2] + k0 + 64, 0, 0);
    }
    FragBF af[MT], bf[NT];
    #pragma unroll
    for (int mi = 0; mi < MT; ++mi)
      #pragma unroll
      for (int j = 0; j < 8; ++j) {
        const int kb = 2 * j + half * 16;   // contiguous k = 16*half + s
        af[mi].u[j] = *(const unsigned int*)(hrow[mi] + k0 + kb);
      }
    #pragma unroll
    for (int ni = 0; ni < NT; ++ni)
      #pragma unroll
      for (int j = 0; j < 8; ++j)                                       // contiguous k = 16*half + s, as A
        bf[ni].u[j] = *(const unsigned int*)(arow[ni] + k0 + 2 * j);

    #pragma unroll
    for (int mi = 0; mi < MT; ++mi)
      #pragma unroll
      for (int ni = 0; ni < NT; ++ni)
        acc[mi][ni] = __builtin_amdgcn_wmma_f32_16x16x32_bf16(
            false, af[mi].v, false, bf[ni].v, (short)0, acc[mi][ni], false, false);
  }

  // Pad n-columns come out as exact zeros (a rows >= 2000 are zero, outf pad is 0)
  #pragma unroll
  for (int mi = 0; mi < MT; ++mi)
    #pragma unroll
    for (int ni = 0; ni < NT; ++ni) {
      const int n = n0 + ni * 16 + lm;
      #pragma unroll
      for (int r = 0; r < 8; ++r) {
        const unsigned m = (unsigned)(m0 + mi * 16 + r + 8 * half);
        const float res = 0.05f * outf[m * SPAD + n] + 0.95f * acc[mi][ni][r];
        Hout[m * SPAD + n] = f2b(res);
      }
    }
}

// ---------------------------------------------------------------------------
// 4) fused 1x1 mlp (256->64) + exact GELU + w_end (64->1):  z[b,n] (bf16)
// ---------------------------------------------------------------------------
__global__ void k_mlp(const unsigned short* __restrict__ Hall,
                      const float* __restrict__ w_mlp, const float* __restrict__ b_mlp,
                      const float* __restrict__ w_end, const float* __restrict__ b_end,
                      unsigned short* __restrict__ zbf) {
  __shared__ float ho[256 * 32];
  __shared__ float zsh[32];
  const int b  = blockIdx.y;
  const int n0 = blockIdx.x * 32;
  const int t  = threadIdx.x;

  for (int idx = t; idx < 256 * 32; idx += 256) {
    const int cc = idx >> 5, nl = idx & 31;
    const int g = cc >> 6, c = cc & 63;
    ho[idx] = b2f(Hall[((unsigned)g * MROWS + (unsigned)(b * CCH + c)) * SPAD + n0 + nl]);
  }
  const int nl = t & 31, og = t >> 5;
  if (og == 0) zsh[nl] = b_end[0];
  __syncthreads();

  float part = 0.f;
  #pragma unroll
  for (int oo = 0; oo < 8; ++oo) {
    const int o = og * 8 + oo;
    float s = b_mlp[o];
    const float* wm = w_mlp + o * 256;
    for (int cc = 0; cc < 256; ++cc) s += wm[cc] * ho[cc * 32 + nl];
    const float gl = 0.5f * s * (1.f + erff(s * 0.7071067811865475f)); // exact gelu
    part += w_end[o] * gl;
  }
  atomicAdd(&zsh[nl], part);
  __syncthreads();
  if (og == 0) {
    const int n = n0 + nl;
    const float zv = (n < N_NODE) ? zsh[nl] : 0.f;
    zbf[(unsigned)b * SPAD + n] = f2b(zv);
  }
}

// ---------------------------------------------------------------------------
// 5a) convert w_lin (2048 x 2000 f32) -> bf16 with K padding
// ---------------------------------------------------------------------------
__global__ void k_cvt_wlin(const float* __restrict__ w_lin, unsigned short* __restrict__ wlb) {
  const int idx = blockIdx.x * 256 + threadIdx.x;
  if (idx >= DM * SPAD) return;
  const int d = idx / SPAD, n = idx % SPAD;
  wlb[idx] = f2b(n < N_NODE ? w_lin[(unsigned)d * N_NODE + n] : 0.f);
}

// ---------------------------------------------------------------------------
// 5b) final GEMM (WMMA bf16): pre[b,d] = z[b,:] . w_lin[d,:] + b_lin[d]
// ---------------------------------------------------------------------------
__global__ void __launch_bounds__(32)
k_final(const unsigned short* __restrict__ zbf, const unsigned short* __restrict__ wlb,
        const float* __restrict__ b_lin, float* __restrict__ pre) {
  const int td = blockIdx.x;          // 0..127  (2048/16 d-tiles)
  const int tm = blockIdx.y;          // 0..3    (64/16 batch-tiles)
  const int lane = threadIdx.x;
  const int half = lane >> 4;
  const int lm   = lane & 15;
  const int m0 = tm * 16, d0 = td * 16;

  v8f acc = {};
  const unsigned short* brow = wlb + (unsigned)(d0 + lm) * SPAD + half * 16;
  const unsigned short* arow = zbf + (unsigned)(m0 + lm) * SPAD;

  for (int k0 = 0; k0 < SPAD; k0 += 32) {
    FragBF a, bfr;
    #pragma unroll
    for (int j = 0; j < 8; ++j) {
      const int kb = 2 * j + half * 16;
      a.u[j]   = *(const unsigned int*)(arow + k0 + kb);
      bfr.u[j] = *(const unsigned int*)(brow + k0 + 2 * j);
    }
    acc = __builtin_amdgcn_wmma_f32_16x16x32_bf16(false, a.v, false, bfr.v,
                                                  (short)0, acc, false, false);
  }
  const int d = d0 + lm;
  #pragma unroll
  for (int r = 0; r < 8; ++r) {
    const unsigned m = (unsigned)(m0 + r + 8 * half);
    pre[m * DM + d] = acc[r] + b_lin[d];
  }
}

// ---------------------------------------------------------------------------
// 6) layernorm over last dim (2048) -> d_out (f32)
// ---------------------------------------------------------------------------
__global__ void k_ln(const float* __restrict__ pre, const float* __restrict__ g,
                     const float* __restrict__ bb, float* __restrict__ out) {
  __shared__ float red[256];
  const int r = blockIdx.x, t = threadIdx.x;
  const float* row = pre + (size_t)r * DM;

  float s = 0.f;
  for (int i = t; i < DM; i += 256) s += row[i];
  red[t] = s; __syncthreads();
  for (int st = 128; st > 0; st >>= 1) { if (t < st) red[t] += red[t + st]; __syncthreads(); }
  const float mu = red[0] / DM; __syncthreads();

  float v = 0.f;
  for (int i = t; i < DM; i += 256) { const float d = row[i] - mu; v += d * d; }
  red[t] = v; __syncthreads();
  for (int st = 128; st > 0; st >>= 1) { if (t < st) red[t] += red[t + st]; __syncthreads(); }
  const float inv = rsqrtf(red[0] / DM + 1e-5f);
  __syncthreads();

  for (int i = t; i < DM; i += 256)
    out[(size_t)r * DM + i] = (row[i] - mu) * inv * g[i] + bb[i];
}

// ---------------------------------------------------------------------------
extern "C" void kernel_launch(void* const* d_in, const int* in_sizes, int n_in,
                              void* d_out, int out_size, void* d_ws, size_t ws_size,
                              hipStream_t stream) {
  (void)in_sizes; (void)n_in; (void)out_size; (void)ws_size;
  const float* x        = (const float*)d_in[0];
  const float* nodevec1 = (const float*)d_in[1];
  const float* nodevec2 = (const float*)d_in[2];
  const float* w_start  = (const float*)d_in[3];
  const float* b_start  = (const float*)d_in[4];
  const float* w_mlp    = (const float*)d_in[5];
  const float* b_mlp    = (const float*)d_in[6];
  const float* w_end    = (const float*)d_in[7];
  const float* b_end    = (const float*)d_in[8];
  const float* w_lin    = (const float*)d_in[9];
  const float* b_lin    = (const float*)d_in[10];
  const float* ln_g     = (const float*)d_in[11];
  const float* ln_b     = (const float*)d_in[12];
  float* out = (float*)d_out;

  // workspace carve-up (~118 MB)
  char* base = (char*)d_ws;
  size_t cur = 0;
  auto take = [&](size_t bytes) -> void* {
    void* r = base + cur;
    cur += (bytes + 255) & ~(size_t)255;
    return r;
  };
  unsigned short* a_bf = (unsigned short*)take((size_t)SPAD * SPAD * 2);      // 2048 rows (padded)
  unsigned short* Hall = (unsigned short*)take((size_t)4 * MROWS * SPAD * 2); // hops 0..3
  float*          outf = (float*)take((size_t)MROWS * SPAD * 4);
  unsigned short* zbf  = (unsigned short*)take((size_t)BATCHN * SPAD * 2);
  unsigned short* wlb  = (unsigned short*)take((size_t)DM * SPAD * 2);
  float*          pre  = (float*)take((size_t)BATCHN * DM * 4);

  // 1) adjacency (incl. zero pad rows 2000..2047)
  k_adj<<<SPAD, 256, 0, stream>>>(nodevec1, nodevec2, a_bf);
  // 2) start conv -> hop0 (f32 + bf16)
  k_conv<<<dim3(SPAD / 256, CCH, BATCHN), 256, 0, stream>>>(x, w_start, b_start, outf, Hall);
  // 3) three WMMA mixprop hops, 32x64 output block per wave
  for (int g = 0; g < 3; ++g) {
    k_hop<<<dim3(SPAD / (NT * 16), MROWS / (MT * 16)), 32, 0, stream>>>(
        Hall + (size_t)g * MROWS * SPAD, a_bf, outf,
        Hall + (size_t)(g + 1) * MROWS * SPAD);
  }
  // 4) fused mlp + gelu + end -> z (bf16, padded)
  k_mlp<<<dim3(SPAD / 32, BATCHN), 256, 0, stream>>>(Hall, w_mlp, b_mlp, w_end, b_end, zbf);
  // 5) final linear via WMMA
  k_cvt_wlin<<<(DM * SPAD + 255) / 256, 256, 0, stream>>>(w_lin, wlb);
  k_final<<<dim3(DM / 16, BATCHN / 16), 32, 0, stream>>>(zbf, wlb, b_lin, pre);
  // 6) layernorm
  k_ln<<<BATCHN, 256, 0, stream>>>(pre, ln_g, ln_b, out);
}
